// CustomMoE_4681514353099
// MI455X (gfx1250) — compile-verified, matrix operand order checked by reading, not run on hardware
//
#include <hip/hip_runtime.h>

// ---------- types ----------
typedef __attribute__((ext_vector_type(16))) __bf16 bfx16;
typedef __attribute__((ext_vector_type(8)))  __bf16 bfx8;
typedef __attribute__((ext_vector_type(4)))  __bf16 bfx4;
typedef __attribute__((ext_vector_type(2)))  __bf16 bfx2;
typedef __attribute__((ext_vector_type(8)))  float  fx8;
typedef __attribute__((ext_vector_type(4)))  float  fx4;
typedef __attribute__((ext_vector_type(4)))  unsigned int ux4;

static constexpr int T_  = 4096;
static constexpr int D_  = 2048;
static constexpr int H_  = 8192;
static constexpr int E_  = 8;
static constexpr int CAP = 2048;   // 2 * ceil(2*4096/8)

__device__ __forceinline__ bfx16 cat8(bfx8 lo, bfx8 hi) {
    return __builtin_shufflevector(lo, hi, 0,1,2,3,4,5,6,7,8,9,10,11,12,13,14,15);
}

// ---------- 1) zero dispatched buffer + counters ----------
__global__ __launch_bounds__(256)
void k_init(ux4* __restrict__ disp4, long n4, int* __restrict__ counters) {
    long i = (long)blockIdx.x * blockDim.x + threadIdx.x;
    long stride = (long)gridDim.x * blockDim.x;
    ux4 z = {0u, 0u, 0u, 0u};
    for (; i < n4; i += stride) disp4[i] = z;
    if (blockIdx.x == 0 && threadIdx.x < 24) counters[threadIdx.x] = 0;
}

// ---------- 2) routing: logits, softmax, top-2, positions, l_aux stats ----------
__global__ __launch_bounds__(256)
void k_route(const float* __restrict__ x, const float* __restrict__ wg,
             int* __restrict__ expert_of, int* __restrict__ pos_of,
             float* __restrict__ gate_of,
             int* __restrict__ counts, int* __restrict__ ce_cnt,
             float* __restrict__ me_sum) {
    const int t = blockIdx.x;
    __shared__ float red[256 * E_];
    float part[E_];
    #pragma unroll
    for (int e = 0; e < E_; ++e) part[e] = 0.f;

    const float* xr = x + (long)t * D_;
    for (int i = threadIdx.x; i < D_; i += 256) {
        float xv = xr[i];
        const float* w = wg + (long)i * E_;
        #pragma unroll
        for (int e = 0; e < E_; ++e) part[e] += xv * w[e];
    }
    #pragma unroll
    for (int e = 0; e < E_; ++e) red[threadIdx.x * E_ + e] = part[e];
    __syncthreads();
    for (int s = 128; s > 0; s >>= 1) {
        if (threadIdx.x < s) {
            #pragma unroll
            for (int e = 0; e < E_; ++e)
                red[threadIdx.x * E_ + e] += red[(threadIdx.x + s) * E_ + e];
        }
        __syncthreads();
    }
    if (threadIdx.x == 0) {
        float lg[E_], sc[E_];
        #pragma unroll
        for (int e = 0; e < E_; ++e) lg[e] = red[e];
        float mx = lg[0];
        #pragma unroll
        for (int e = 1; e < E_; ++e) mx = fmaxf(mx, lg[e]);
        float se = 0.f;
        #pragma unroll
        for (int e = 0; e < E_; ++e) { sc[e] = __expf(lg[e] - mx); se += sc[e]; }
        float inv = 1.f / se;
        #pragma unroll
        for (int e = 0; e < E_; ++e) sc[e] *= inv;

        int i0 = 0;
        #pragma unroll
        for (int e = 1; e < E_; ++e) if (sc[e] > sc[i0]) i0 = e;
        int i1 = (i0 == 0) ? 1 : 0;
        #pragma unroll
        for (int e = 0; e < E_; ++e) if (e != i0 && sc[e] > sc[i1]) i1 = e;

        #pragma unroll
        for (int e = 0; e < E_; ++e) atomicAdd(&me_sum[e], sc[e]);
        atomicAdd(&ce_cnt[i0], 1);
        int p0 = atomicAdd(&counts[i0], 1);
        int p1 = atomicAdd(&counts[i1], 1);
        expert_of[t]       = i0;  pos_of[t]       = p0;  gate_of[t]       = sc[i0];
        expert_of[T_ + t]  = i1;  pos_of[T_ + t]  = p1;  gate_of[T_ + t]  = sc[i1];
    }
}

// ---------- 3) dispatch: scatter x rows (f32 -> bf16) into [E, CAP, D] ----------
__global__ __launch_bounds__(256)
void k_dispatch(const float* __restrict__ x, const int* __restrict__ expert_of,
                const int* __restrict__ pos_of, __bf16* __restrict__ disp) {
    const int s = blockIdx.x;             // slot-major: s<T -> slot0 token s; else slot1
    const int p = pos_of[s];
    if (p >= CAP) return;
    const int e = expert_of[s];
    const int t = s & (T_ - 1);
    const float* src = x + (long)t * D_;
    __bf16* dst = disp + ((long)e * CAP + p) * D_;
    for (int i = threadIdx.x * 4; i < D_; i += 256 * 4) {
        fx4 v = *(const fx4*)(src + i);
        bfx4 o;
        o[0] = (__bf16)v[0]; o[1] = (__bf16)v[1];
        o[2] = (__bf16)v[2]; o[3] = (__bf16)v[3];
        *(bfx4*)(dst + i) = o;
    }
}

// ---------- 4) WMMA GEMM: per-expert  Out[e] = act(A[e] x B[e] + bias[e]) ----------
// A: bf16 [M,K] row-major; B: f32 [K,N] row-major (converted to packed bf16 pairs
// in registers, transposed into LDS as [n][k]).
// Block tile 128x128, 4 waves (64x64 each), K-tile 32, double-buffered LDS with
// register prefetch of the next tile issued before the current tile's WMMAs.
template <bool RELU_TO_BF16>
__global__ __launch_bounds__(128)
void k_gemm(const __bf16* __restrict__ A, const float* __restrict__ B,
            const float* __restrict__ bias, void* __restrict__ Out,
            int M, int N, int Kd) {
    constexpr int LDT = 40;                    // bf16 elements per LDS row (pad vs 32)
    __shared__ __bf16 Als[2][128 * LDT];       // [m][k]
    __shared__ __bf16 Bls[2][128 * LDT];       // [n][k]  (transposed while staging)

    const int e     = blockIdx.z;
    const int bm    = blockIdx.y * 128;
    const int bn    = blockIdx.x * 128;
    const int tid   = threadIdx.x;
    const int lane  = tid & 31;
    const int wave  = tid >> 5;
    const int wm    = (wave >> 1) * 64;
    const int wn    = (wave & 1) * 64;
    const int r16   = lane & 15;
    const int half  = lane >> 4;
    const int n4    = lane * 4;                // B-stage column group
    const int kbase = wave * 8;                // B-stage k range [kbase, kbase+8)

    const __bf16* Ae = A + (long)e * M * Kd;
    const float*  Be = B + (long)e * Kd * N;
    const __bf16* Arow = Ae + (long)(bm + tid) * Kd;

    fx8 acc[4][4];
    #pragma unroll
    for (int i = 0; i < 4; ++i)
        #pragma unroll
        for (int j = 0; j < 4; ++j)
            #pragma unroll
            for (int q = 0; q < 8; ++q) acc[i][j][q] = 0.f;

    ux4  areg[4];          // 64B of A row (32 bf16)
    bfx2 breg[4][4];       // [k-pair][col] packed bf16 pairs (k, k+1)

    auto load_tile = [&](int kt) {
        const __bf16* asrc = Arow + kt * 32;
        #pragma unroll
        for (int j = 0; j < 4; ++j) areg[j] = *(const ux4*)(asrc + j * 8);
        #pragma unroll
        for (int kk2 = 0; kk2 < 4; ++kk2) {
            const int k0 = kbase + kk2 * 2;
            const float* b0 = Be + (long)(kt * 32 + k0) * N + bn + n4;
            fx4 v0 = *(const fx4*)b0;
            fx4 v1 = *(const fx4*)(b0 + N);
            #pragma unroll
            for (int c = 0; c < 4; ++c) {
                bfx2 p;
                p[0] = (__bf16)v0[c];
                p[1] = (__bf16)v1[c];
                breg[kk2][c] = p;
            }
        }
    };

    auto store_tile = [&](int buf) {
        __bf16* adst = &Als[buf][tid * LDT];
        #pragma unroll
        for (int j = 0; j < 4; ++j) *(ux4*)(adst + j * 8) = areg[j];
        #pragma unroll
        for (int kk2 = 0; kk2 < 4; ++kk2)
            #pragma unroll
            for (int c = 0; c < 4; ++c)
                *(bfx2*)(&Bls[buf][(n4 + c) * LDT + kbase + kk2 * 2]) = breg[kk2][c];
    };

    auto compute = [&](int buf) {
        bfx16 af[4], bfr[4];
        #pragma unroll
        for (int i = 0; i < 4; ++i) {
            const __bf16* p = &Als[buf][(wm + i * 16 + r16) * LDT + half * 8];
            af[i] = cat8(*(const bfx8*)p, *(const bfx8*)(p + 16));
        }
        #pragma unroll
        for (int j = 0; j < 4; ++j) {
            const __bf16* p = &Bls[buf][(wn + j * 16 + r16) * LDT + half * 8];
            bfr[j] = cat8(*(const bfx8*)p, *(const bfx8*)(p + 16));
        }
        #pragma unroll
        for (int i = 0; i < 4; ++i)
            #pragma unroll
            for (int j = 0; j < 4; ++j)
                acc[i][j] = __builtin_amdgcn_wmma_f32_16x16x32_bf16(
                    false, af[i], false, bfr[j], (short)0, acc[i][j], false, false);
    };

    const int kTiles = Kd >> 5;
    // prologue: stage tile 0
    load_tile(0);
    store_tile(0);
    __syncthreads();

    for (int kt = 0; kt < kTiles; ++kt) {
        const int cur = kt & 1;
        const bool hasNext = (kt + 1) < kTiles;
        if (hasNext) load_tile(kt + 1);   // global loads in flight during WMMAs
        compute(cur);
        if (hasNext) {
            __syncthreads();              // all waves done reading buf cur^1
            store_tile(cur ^ 1);          // waits on the prefetch loads here
            __syncthreads();
        }
    }

    // epilogue: bias (+ReLU) ; C-frag layout: m_local = half*8 + v, n_local = r16
    const float* be = bias + (long)e * N;
    #pragma unroll
    for (int j = 0; j < 4; ++j) {
        const int n = bn + wn + j * 16 + r16;
        const float bv = be[n];
        #pragma unroll
        for (int i = 0; i < 4; ++i) {
            #pragma unroll
            for (int v = 0; v < 8; ++v) {
                const int m = bm + wm + i * 16 + half * 8 + v;
                float val = acc[i][j][v] + bv;
                const long idx = (long)e * M * N + (long)m * N + n;
                if (RELU_TO_BF16) {
                    val = val > 0.f ? val : 0.f;
                    ((__bf16*)Out)[idx] = (__bf16)val;
                } else {
                    ((float*)Out)[idx] = val;
                }
            }
        }
    }
}

// ---------- 5) combine: out[t] = g0*y[e0,p0] + g1*y[e1,p1] ----------
__global__ __launch_bounds__(256)
void k_combine(const float* __restrict__ y, const int* __restrict__ expert_of,
               const int* __restrict__ pos_of, const float* __restrict__ gate_of,
               float* __restrict__ out) {
    const int t = blockIdx.x;
    int e0 = expert_of[t],      p0 = pos_of[t];
    int e1 = expert_of[T_ + t], p1 = pos_of[T_ + t];
    float g0 = (p0 < CAP) ? gate_of[t] : 0.f;
    float g1 = (p1 < CAP) ? gate_of[T_ + t] : 0.f;
    if (p0 >= CAP) p0 = 0;
    if (p1 >= CAP) p1 = 0;
    const float* r0 = y + ((long)e0 * CAP + p0) * D_;
    const float* r1 = y + ((long)e1 * CAP + p1) * D_;
    float* o = out + (long)t * D_;
    for (int i = threadIdx.x * 4; i < D_; i += 256 * 4) {
        fx4 a = *(const fx4*)(r0 + i);
        fx4 b = *(const fx4*)(r1 + i);
        fx4 r;
        #pragma unroll
        for (int c = 0; c < 4; ++c) r[c] = g0 * a[c] + g1 * b[c];
        *(fx4*)(o + i) = r;
    }
}

// ---------- 6) l_aux = E * sum_e (me_sum[e]/T) * (ce_cnt[e]/T) ----------
__global__ void k_laux(const float* __restrict__ me_sum, const int* __restrict__ ce_cnt,
                       float* __restrict__ out_scalar) {
    if (threadIdx.x == 0 && blockIdx.x == 0) {
        float s = 0.f;
        #pragma unroll
        for (int e = 0; e < E_; ++e) s += me_sum[e] * (float)ce_cnt[e];
        out_scalar[0] = (float)E_ * s / ((float)T_ * (float)T_);
    }
}

// ---------- launch ----------
extern "C" void kernel_launch(void* const* d_in, const int* in_sizes, int n_in,
                              void* d_out, int out_size, void* d_ws, size_t ws_size,
                              hipStream_t stream) {
    const float* x  = (const float*)d_in[0];
    const float* wg = (const float*)d_in[1];
    const float* w1 = (const float*)d_in[2];
    const float* b1 = (const float*)d_in[3];
    const float* w2 = (const float*)d_in[4];
    const float* b2 = (const float*)d_in[5];
    float* out = (float*)d_out;

    char* ws = (char*)d_ws;
    size_t off = 0;
    __bf16* disp = (__bf16*)(ws + off); off += (size_t)E_ * CAP * D_ * 2;   // 67 MB
    __bf16* hbuf = (__bf16*)(ws + off); off += (size_t)E_ * CAP * H_ * 2;   // 268 MB
    float*  ybuf = (float*) (ws + off); off += (size_t)E_ * CAP * D_ * 4;   // 134 MB
    int*   expert_of = (int*)  (ws + off); off += (size_t)2 * T_ * 4;
    int*   pos_of    = (int*)  (ws + off); off += (size_t)2 * T_ * 4;
    float* gate_of   = (float*)(ws + off); off += (size_t)2 * T_ * 4;
    int*   counters  = (int*)  (ws + off);   // counts[8], ce[8], me[8](f32)
    int*   counts = counters;
    int*   ce     = counters + 8;
    float* me     = (float*)(counters + 16);

    const long disp4 = (long)E_ * CAP * D_ * 2 / 16;
    k_init<<<4096, 256, 0, stream>>>((ux4*)disp, disp4, counters);
    k_route<<<T_, 256, 0, stream>>>(x, wg, expert_of, pos_of, gate_of, counts, ce, me);
    k_dispatch<<<2 * T_, 256, 0, stream>>>(x, expert_of, pos_of, disp);
    k_gemm<true ><<<dim3(H_ / 128, CAP / 128, E_), 128, 0, stream>>>(
        disp, w1, b1, (void*)hbuf, CAP, H_, D_);
    k_gemm<false><<<dim3(D_ / 128, CAP / 128, E_), 128, 0, stream>>>(
        hbuf, w2, b2, (void*)ybuf, CAP, D_, H_);
    k_combine<<<T_, 256, 0, stream>>>(ybuf, expert_of, pos_of, gate_of, out);
    k_laux<<<1, 32, 0, stream>>>(me, ce, out + (size_t)T_ * D_);
}
